// MHSA_11596411699410
// MI455X (gfx1250) — matrix operand order repaired and checked by reading, not compiled, not run on hardware
//
#include <hip/hip_runtime.h>

typedef __attribute__((ext_vector_type(16))) _Float16     v16h;
typedef __attribute__((ext_vector_type(8)))  float        v8f;
typedef __attribute__((ext_vector_type(4)))  unsigned int u32x4;
typedef __attribute__((ext_vector_type(8)))  int          i32x8;
typedef __attribute__((ext_vector_type(4)))  int          i32x4;

#define B_    4
#define C_    256
#define N_    2304        // 48*48 tokens per batch
#define NH_   8
#define DH_   32
#define BT_   (B_ * N_)   // 9216 total tokens
#define C3_   (3 * C_)    // 768
#define SCALE_ 0.17677669529663687f   // dh^-0.5, dh = 32

#if defined(__has_builtin)
#if __has_builtin(__builtin_amdgcn_tensor_load_to_lds) && \
    __has_builtin(__builtin_amdgcn_s_wait_tensorcnt)
#define HAVE_TDM 1
#endif
#endif
#ifndef HAVE_TDM
#define HAVE_TDM 0
#endif

#if HAVE_TDM
// ---------------------------------------------------------------------------
// Issue one TDM tile load (Global -> LDS), D# built per ISA 08_async_tensor:
//   group0: count=1, lds_addr, global_addr[56:0], type=2
//   group1: data_size=2B, tensor dims (large, OOB-safe), tile0/tile1,
//           tensor_dim0_stride (48-bit, data_size units)
// tile0 = elements per row, tile1 = rows, stride0 = row pitch in elements.
// ---------------------------------------------------------------------------
__device__ static inline void tdm_load_tile(unsigned int lds_off, const void* gptr,
                                            unsigned int tile0, unsigned int tile1,
                                            unsigned long long stride0)
{
    unsigned long long ga = (unsigned long long)(uintptr_t)gptr;
    u32x4 g0;
    g0[0] = 1u;                                           // count=1, user mode
    g0[1] = lds_off;                                      // LDS byte address
    g0[2] = (unsigned int)ga;                             // global_addr[31:0]
    g0[3] = (unsigned int)((ga >> 32) & 0x01FFFFFFu) | 0x80000000u; // [56:32]+type=2

    const unsigned long long td0 = 1ull << 20;            // tensor dims: large
    const unsigned long long td1 = 1ull << 20;            //  (no OOB clipping)
    unsigned long long q0 = 0x10000ull                    // data_size=1 (2 bytes)
                          | ((td0 & 0xFFFFull) << 48);    // tensor_dim0[15:0]
    unsigned long long q1 = (td0 >> 16)                   // tensor_dim0[31:16]
                          | (td1 << 16)                   // tensor_dim1
                          | ((unsigned long long)tile0 << 48);  // tile_dim0
    unsigned long long q2 = (unsigned long long)tile1     // tile_dim1 (tile2=0)
                          | ((stride0 & 0xFFFFFFFFull) << 32);  // dim0_stride[31:0]
    unsigned long long q3 = (stride0 >> 32) & 0xFFFFull;  // dim0_stride[47:32]

    i32x8 g1;
    g1[0] = (int)q0; g1[1] = (int)(q0 >> 32);
    g1[2] = (int)q1; g1[3] = (int)(q1 >> 32);
    g1[4] = (int)q2; g1[5] = (int)(q2 >> 32);
    g1[6] = (int)q3; g1[7] = (int)(q3 >> 32);
    i32x4 z4 = {0, 0, 0, 0};
#if __clang_major__ >= 23
    i32x8 z8 = {0, 0, 0, 0, 0, 0, 0, 0};
    __builtin_amdgcn_tensor_load_to_lds(g0, g1, z4, z4, z8, 0);
#else
    __builtin_amdgcn_tensor_load_to_lds(g0, g1, z4, z4, 0);
#endif
}
#endif

// ---------------------------------------------------------------------------
// Kernel 0: one-shot fp32 -> f16 conversion of both weight matrices.
// ---------------------------------------------------------------------------
__global__ __launch_bounds__(256)
void cvt_kernel(const float* __restrict__ wq, const float* __restrict__ wp,
                _Float16* __restrict__ wqh, _Float16* __restrict__ wph)
{
    int i = blockIdx.x * 256 + threadIdx.x;
    if (i < C3_ * C_) wqh[i] = (_Float16)wq[i];
    if (i < C_ * C_)  wph[i] = (_Float16)wp[i];
}

// ---------------------------------------------------------------------------
// Kernel 1: fused QKV projection (16 tokens x 32 outputs per wave, x tile
// staged in LDS). Q pre-scaled by dh^-0.5; Q,K stored [B][NH][N][DH];
// V stored transposed [B][NH][DH][N].
// ---------------------------------------------------------------------------
__global__ __launch_bounds__(256)
void qkv_kernel(const float* __restrict__ x, const _Float16* __restrict__ wh,
                const float* __restrict__ bias,
                _Float16* __restrict__ Qb, _Float16* __restrict__ Kb,
                _Float16* __restrict__ Vt)
{
    __shared__ _Float16 xs[16][C_];                 // 8 KB
    const int tid = threadIdx.x;
    const int bt0 = blockIdx.x * 16;
    const int b   = bt0 / N_;
    const int n0  = bt0 - b * N_;

    for (int idx = tid; idx < 16 * C_; idx += 256) {
        int t = idx & 15;
        int c = idx >> 4;
        xs[t][c] = (_Float16)x[(size_t)b * C_ * N_ + (size_t)c * N_ + (n0 + t)];
    }
    __syncthreads();

    const int wave  = tid >> 5;
    const int lane  = tid & 31;
    const int nn    = lane & 15;
    const int hi    = lane >> 4;
    const int kbA   = hi ? 16 : 0;
    const int kbB   = hi ? 16 : 0;
    const int obase = blockIdx.y * 256 + wave * 32;

    const _Float16* wrow0 = wh + (size_t)(obase + nn) * C_;
    const _Float16* wrow1 = wh + (size_t)(obase + 16 + nn) * C_;
    v8f acc0 = {}, acc1 = {};
    for (int k0 = 0; k0 < C_; k0 += 32) {
        v16h a, bm0, bm1;
        #pragma unroll
        for (int i = 0; i < 8; ++i) {
            a[i]     = xs[nn][k0 + kbA + i];
            a[8 + i] = xs[nn][k0 + kbA + 8 + i];
        }
        #pragma unroll
        for (int i = 0; i < 16; ++i) {
            bm0[i] = wrow0[k0 + kbB + i];
            bm1[i] = wrow1[k0 + kbB + i];
        }
        acc0 = __builtin_amdgcn_wmma_f32_16x16x32_f16(false, a, false, bm0,
                                                      (short)0, acc0, false, false);
        acc1 = __builtin_amdgcn_wmma_f32_16x16x32_f16(false, a, false, bm1,
                                                      (short)0, acc1, false, false);
    }

    const float bo0 = bias[obase + nn];
    const float bo1 = bias[obase + 16 + nn];
    const int which = obase >> 8;
    const int head  = (obase & 255) >> 5;
    const size_t base = ((size_t)(b * NH_ + head)) * N_ * DH_;
    #pragma unroll
    for (int r = 0; r < 8; ++r) {
        int tok = n0 + r + (hi ? 8 : 0);
        float v0 = acc0[r] + bo0;
        float v1 = acc1[r] + bo1;
        if (which == 0) {
            Qb[base + (size_t)tok * DH_ + nn]       = (_Float16)(v0 * SCALE_);
            Qb[base + (size_t)tok * DH_ + 16 + nn]  = (_Float16)(v1 * SCALE_);
        } else if (which == 1) {
            Kb[base + (size_t)tok * DH_ + nn]       = (_Float16)v0;
            Kb[base + (size_t)tok * DH_ + 16 + nn]  = (_Float16)v1;
        } else {
            Vt[base + (size_t)nn * N_ + tok]        = (_Float16)v0;
            Vt[base + (size_t)(16 + nn) * N_ + tok] = (_Float16)v1;
        }
    }
}

// ---------------------------------------------------------------------------
// Kernel 2: flash attention, one wave per (batch, head, 16-query tile).
// K (2 KB contiguous) and V^T (32x32, stride N) tiles are double-buffered in
// LDS via the Tensor Data Mover: issue tile t+1, s_wait_tensorcnt(2) for
// tile t, compute (2 WMMAs S, online softmax, 2 WMMAs PV).
// ---------------------------------------------------------------------------
__global__ __launch_bounds__(32)
void attn_kernel(const _Float16* __restrict__ Qb, const _Float16* __restrict__ Kb,
                 const _Float16* __restrict__ Vt, _Float16* __restrict__ Ob)
{
    __shared__ _Float16 lds_k[2][32][32];           // 4 KB double buffer
    __shared__ _Float16 lds_v[2][32][32];           // 4 KB double buffer
    __shared__ _Float16 ps[16][32];                 // 1 KB P bounce
    const int lane = threadIdx.x;
    const int nn   = lane & 15;
    const int hi   = lane >> 4;
    const int kbA  = hi ? 16 : 0;
    const int kbB  = hi ? 16 : 0;
    const int q0   = blockIdx.x * 16;
    const int head = blockIdx.y;
    const int b    = blockIdx.z;
    const size_t base = ((size_t)(b * NH_ + head)) * N_ * DH_;
    const _Float16* kbase = Kb + base;
    const _Float16* vbase = Vt + base;

    // Q tile as WMMA A-operand (16 x 32); dh==32 -> single K step.
    v16h aq;
    const _Float16* qrow = Qb + base + (size_t)(q0 + nn) * DH_;
    #pragma unroll
    for (int i = 0; i < 8; ++i) { aq[i] = qrow[kbA + i]; aq[8 + i] = qrow[kbA + 8 + i]; }

#if HAVE_TDM
    const unsigned int koff0 = (unsigned int)(uintptr_t)&lds_k[0][0][0];
    const unsigned int koff1 = (unsigned int)(uintptr_t)&lds_k[1][0][0];
    const unsigned int voff0 = (unsigned int)(uintptr_t)&lds_v[0][0][0];
    const unsigned int voff1 = (unsigned int)(uintptr_t)&lds_v[1][0][0];
    // Prime buffer 0 with the first 32-key tile.
    tdm_load_tile(koff0, kbase, 1024, 1, 1024);           // 32x32 f16 contiguous
    tdm_load_tile(voff0, vbase, 32, 32, (unsigned long long)N_); // 32 rows, pitch N
#endif

    float mrow[8], lrow[8];
    #pragma unroll
    for (int r = 0; r < 8; ++r) { mrow[r] = -3.0e38f; lrow[r] = 0.0f; }
    v8f o0 = {}, o1 = {};

    for (int kt = 0; kt < N_; kt += 32) {
        const int buf = (kt >> 5) & 1;
        v16h bk0, bk1, bv0, bv1;
#if HAVE_TDM
        if (kt + 32 < N_) {                         // prefetch next tile via TDM
            const unsigned int nk = buf ? koff0 : koff1;
            const unsigned int nv = buf ? voff0 : voff1;
            tdm_load_tile(nk, kbase + (size_t)(kt + 32) * DH_, 1024, 1, 1024);
            tdm_load_tile(nv, vbase + (kt + 32), 32, 32, (unsigned long long)N_);
            __builtin_amdgcn_s_wait_tensorcnt(2);   // current pair has landed
        } else {
            __builtin_amdgcn_s_wait_tensorcnt(0);
        }
        // TDM writes LDS behind the compiler's back: fence reordering.
        asm volatile("" : : "r"(&lds_k[0][0][0]), "r"(&lds_v[0][0][0]) : "memory");
        #pragma unroll
        for (int i = 0; i < 16; ++i) {
            bk0[i] = lds_k[buf][nn][kbB + i];       // B[c][key] = K[key][c]
            bk1[i] = lds_k[buf][16 + nn][kbB + i];
            bv0[i] = lds_v[buf][nn][kbB + i];       // B[k][n]  = Vt[n][kt+k]
            bv1[i] = lds_v[buf][16 + nn][kbB + i];
        }
#else
        (void)buf;
        const _Float16* k0r = kbase + (size_t)(kt + nn) * DH_;
        const _Float16* k1r = kbase + (size_t)(kt + 16 + nn) * DH_;
        const _Float16* v0r = vbase + (size_t)nn * N_ + kt;
        const _Float16* v1r = vbase + (size_t)(16 + nn) * N_ + kt;
        #pragma unroll
        for (int i = 0; i < 16; ++i) {
            bk0[i] = k0r[kbB + i];
            bk1[i] = k1r[kbB + i];
            bv0[i] = v0r[kbB + i];
            bv1[i] = v1r[kbB + i];
        }
#endif
        // S = (Q*scale) * K^T for 32 keys: two 16x16 f32 tiles.
        v8f z = {};
        v8f s0 = __builtin_amdgcn_wmma_f32_16x16x32_f16(false, aq, false, bk0, (short)0, z, false, false);
        v8f s1 = __builtin_amdgcn_wmma_f32_16x16x32_f16(false, aq, false, bk1, (short)0, z, false, false);

        __syncthreads();  // single-wave WG: orders ps reuse across iterations
        #pragma unroll
        for (int r = 0; r < 8; ++r) {
            float x0 = s0[r];
            float x1 = s1[r];
            float mx = fmaxf(x0, x1);
            #pragma unroll
            for (int sh = 1; sh < 16; sh <<= 1)     // xor 1/2/4/8 stays in 16-group
                mx = fmaxf(mx, __shfl_xor(mx, sh, 32));
            float mnew = fmaxf(mrow[r], mx);
            float p0 = __expf(x0 - mnew);
            float p1 = __expf(x1 - mnew);
            int row = r + (hi ? 8 : 0);
            ps[row][nn]      = (_Float16)p0;
            ps[row][16 + nn] = (_Float16)p1;
            float psum = p0 + p1;
            #pragma unroll
            for (int sh = 1; sh < 16; sh <<= 1)
                psum += __shfl_xor(psum, sh, 32);
            float corr = __expf(mrow[r] - mnew);
            lrow[r] = lrow[r] * corr + psum;
            o0[r] *= corr;
            o1[r] *= corr;
            mrow[r] = mnew;
        }
        __syncthreads();

        // P (16x32) back out of LDS in A-operand layout
        v16h ap;
        #pragma unroll
        for (int i = 0; i < 8; ++i) { ap[i] = ps[nn][kbA + i]; ap[8 + i] = ps[nn][kbA + 8 + i]; }

        o0 = __builtin_amdgcn_wmma_f32_16x16x32_f16(false, ap, false, bv0, (short)0, o0, false, false);
        o1 = __builtin_amdgcn_wmma_f32_16x16x32_f16(false, ap, false, bv1, (short)0, o1, false, false);
    }

    // normalize; store f16 [BT][C], channel = head*32 + d (proj-friendly)
    #pragma unroll
    for (int r = 0; r < 8; ++r) {
        int row = r + (hi ? 8 : 0);
        float inv = 1.0f / lrow[r];
        size_t rowoff = ((size_t)(b * N_ + q0 + row)) * C_ + head * DH_;
        Ob[rowoff + nn]      = (_Float16)(o0[r] * inv);
        Ob[rowoff + 16 + nn] = (_Float16)(o1[r] * inv);
    }
}

// ---------------------------------------------------------------------------
// Kernel 3: output projection + bias, written back as [B, C, 48, 48] fp32.
// ---------------------------------------------------------------------------
__global__ __launch_bounds__(256)
void proj_kernel(const _Float16* __restrict__ Ob, const _Float16* __restrict__ wh,
                 const float* __restrict__ bias, float* __restrict__ out)
{
    const int tid   = threadIdx.x;
    const int wave  = tid >> 5;
    const int lane  = tid & 31;
    const int nn    = lane & 15;
    const int hi    = lane >> 4;
    const int kbA   = hi ? 16 : 0;
    const int kbB   = hi ? 16 : 0;
    const int bt0   = blockIdx.x * 16;
    const int obase = wave * 32;

    const _Float16* arow  = Ob + (size_t)(bt0 + nn) * C_;
    const _Float16* wrow0 = wh + (size_t)(obase + nn) * C_;
    const _Float16* wrow1 = wh + (size_t)(obase + 16 + nn) * C_;
    v8f acc0 = {}, acc1 = {};
    for (int k0 = 0; k0 < C_; k0 += 32) {
        v16h a, bm0, bm1;
        #pragma unroll
        for (int i = 0; i < 8; ++i) {
            a[i]     = arow[k0 + kbA + i];
            a[8 + i] = arow[k0 + kbA + 8 + i];
        }
        #pragma unroll
        for (int i = 0; i < 16; ++i) {
            bm0[i] = wrow0[k0 + kbB + i];
            bm1[i] = wrow1[k0 + kbB + i];
        }
        acc0 = __builtin_amdgcn_wmma_f32_16x16x32_f16(false, a, false, bm0,
                                                      (short)0, acc0, false, false);
        acc1 = __builtin_amdgcn_wmma_f32_16x16x32_f16(false, a, false, bm1,
                                                      (short)0, acc1, false, false);
    }
    const float bo0 = bias[obase + nn];
    const float bo1 = bias[obase + 16 + nn];
    const int bb = bt0 / N_;
    const int n0 = bt0 - bb * N_;
    #pragma unroll
    for (int r = 0; r < 8; ++r) {
        int row = r + (hi ? 8 : 0);
        size_t bboff = (size_t)bb * C_ * N_ + (n0 + row);
        out[bboff + (size_t)(obase + nn) * N_]      = acc0[r] + bo0;
        out[bboff + (size_t)(obase + 16 + nn) * N_] = acc1[r] + bo1;
    }
}

extern "C" void kernel_launch(void* const* d_in, const int* in_sizes, int n_in,
                              void* d_out, int out_size, void* d_ws, size_t ws_size,
                              hipStream_t stream)
{
    const float* x      = (const float*)d_in[0];
    const float* w_qkv  = (const float*)d_in[1];
    const float* b_qkv  = (const float*)d_in[2];
    const float* w_proj = (const float*)d_in[3];
    const float* b_proj = (const float*)d_in[4];
    float* out = (float*)d_out;

    _Float16* ws = (_Float16*)d_ws;
    const size_t QS = (size_t)B_ * NH_ * N_ * DH_;  // 2,359,296 elems each
    _Float16* Qb  = ws;
    _Float16* Kb  = ws + QS;
    _Float16* Vt  = ws + 2 * QS;
    _Float16* Ob  = ws + 3 * QS;
    _Float16* wqh = ws + 4 * QS;
    _Float16* wph = wqh + (size_t)C3_ * C_;         // ~19.4 MB total

    cvt_kernel <<<dim3((C3_ * C_ + 255) / 256), 256, 0, stream>>>(w_qkv, w_proj, wqh, wph);
    qkv_kernel <<<dim3(BT_ / 16, C3_ / 256), 256, 0, stream>>>(x, wqh, b_qkv, Qb, Kb, Vt);
    attn_kernel<<<dim3(N_ / 16, NH_, B_),     32, 0, stream>>>(Qb, Kb, Vt, Ob);
    proj_kernel<<<dim3(BT_ / 16),            256, 0, stream>>>(Ob, wph, b_proj, out);
}
